// VPNNetwork_85899346321
// MI455X (gfx1250) — matrix operand order repaired and checked by reading, not compiled, not run on hardware
//
#include <hip/hip_runtime.h>
#include <hip/hip_bf16.h>
#include <math.h>

// ---------------------------------------------------------------------------
// VPN network: phi-GEMM (WMMA f16->f32, double-buffered LDS pipeline) +
// per-image LDS value propagation + agent gather + tiny MLP head.
// B=4096, H=W=32, K=20 steps, A=4 actions.
// ---------------------------------------------------------------------------

typedef __attribute__((ext_vector_type(16))) _Float16 v16h;
typedef __attribute__((ext_vector_type(8)))  _Float16 v8h;
typedef __attribute__((ext_vector_type(4)))  _Float16 v4h;
typedef __attribute__((ext_vector_type(8)))  float    v8f;

#define MDIM 4096
#define NDIM 2048
#define KDIM 3072

#define TILE_M 128
#define TILE_N 128
#define TILE_K 32
#define LDA 40   // padded row stride (halves): 80 B rows -> 16 B aligned frags
#define LDB 40

__device__ __forceinline__ v16h cat8(v8h a, v8h b) {
    return __builtin_shufflevector(a, b, 0,1,2,3,4,5,6,7,8,9,10,11,12,13,14,15);
}

// ---------------------------------------------------------------------------
// Kernel 1: phi[m][n] = sigmoid( obs[m][:] . Phi_w[:][n] + Phi_b[n] )
// Grid: (NDIM/128, MDIM/128), block: 256 threads (8 waves, wave32).
// ---------------------------------------------------------------------------
__global__ __launch_bounds__(256)
void vpn_phi_gemm(const float* __restrict__ obs,     // (4096,3072)
                  const float* __restrict__ phiw,    // (3072,2048)
                  const float* __restrict__ phib,    // (2048)
                  float* __restrict__ phi)           // (4096,2048)
{
    __shared__ _Float16 As[2][TILE_M * LDA];   // [buf][row m][k]
    __shared__ _Float16 Bt[2][TILE_N * LDB];   // [buf][col n][k] (B transposed)

    const int tid  = threadIdx.x;
    const int lane = tid & 31;
    const int wave = tid >> 5;

    const int n0 = blockIdx.x * TILE_N;
    const int m0 = blockIdx.y * TILE_M;

    // 8 waves: 4 along M (32 rows each), 2 along N (64 cols each)
    const int wm = (wave & 3) * 32;
    const int wn = (wave >> 2) * 64;

    const int row16 = lane & 15;
    const int hi    = (lane >> 4) & 1;

    v8f acc[2][4];
#pragma unroll
    for (int ri = 0; ri < 2; ++ri)
#pragma unroll
        for (int ci = 0; ci < 4; ++ci)
            acc[ri][ci] = (v8f){0.f,0.f,0.f,0.f,0.f,0.f,0.f,0.f};

    float4 ar[4];   // staged A tile (this thread's share)
    float4 br[4];   // staged B tile

    auto load_regs = [&](int kk) {
#pragma unroll
        for (int i = 0; i < 4; ++i) {
            int l   = i * 256 + tid;     // 1024 float4 slots
            int row = l >> 3;            // A: 8 float4 per 32-k row
            int k4  = l & 7;
            ar[i] = *reinterpret_cast<const float4*>(
                obs + (size_t)(m0 + row) * KDIM + kk + k4 * 4);
            int krow = l >> 5;           // B: 32 float4 per 128-n k-row
            int n4   = l & 31;
            br[i] = *reinterpret_cast<const float4*>(
                phiw + (size_t)(kk + krow) * NDIM + n0 + n4 * 4);
        }
    };

    auto store_lds = [&](int buf) {
#pragma unroll
        for (int i = 0; i < 4; ++i) {
            int l   = i * 256 + tid;
            int row = l >> 3;
            int k4  = l & 7;
            v4h pa = { (_Float16)ar[i].x, (_Float16)ar[i].y,
                       (_Float16)ar[i].z, (_Float16)ar[i].w };
            *reinterpret_cast<v4h*>(&As[buf][row * LDA + k4 * 4]) = pa;  // b64
            int krow = l >> 5;
            int n4   = l & 31;
            Bt[buf][(n4 * 4 + 0) * LDB + krow] = (_Float16)br[i].x;
            Bt[buf][(n4 * 4 + 1) * LDB + krow] = (_Float16)br[i].y;
            Bt[buf][(n4 * 4 + 2) * LDB + krow] = (_Float16)br[i].z;
            Bt[buf][(n4 * 4 + 3) * LDB + krow] = (_Float16)br[i].w;
        }
    };

    // ---- pipeline prologue ----
    load_regs(0);
    store_lds(0);
    __syncthreads();

    const int KT = KDIM / TILE_K;   // 96
    for (int kt = 0; kt < KT; ++kt) {
        const int cur = kt & 1;
        if (kt + 1 < KT) load_regs((kt + 1) * TILE_K);   // prefetch next tile

        // ---- fragments (16 B aligned -> ds_load_b128) ----
        v16h afrag[2];
#pragma unroll
        for (int ri = 0; ri < 2; ++ri) {
            const _Float16* ap = &As[cur][(wm + ri * 16 + row16) * LDA];
            const int kb = hi ? 8 : 0;   // lanes 16-31 hold K 8..15 / 24..31
            v8h lo = *reinterpret_cast<const v8h*>(ap + kb);        // K kb..kb+7
            v8h hi8 = *reinterpret_cast<const v8h*>(ap + kb + 16);  // K kb+16..kb+23
            afrag[ri] = cat8(lo, hi8);
        }
        v16h bfrag[4];
#pragma unroll
        for (int ci = 0; ci < 4; ++ci) {
            const _Float16* bp = &Bt[cur][(wn + ci * 16 + row16) * LDB];
            const int kb = hi ? 8 : 0;   // lanes 16-31 hold K 8..15 / 24..31
            v8h lo = *reinterpret_cast<const v8h*>(bp + kb);
            v8h hi8 = *reinterpret_cast<const v8h*>(bp + kb + 16);
            bfrag[ci] = cat8(lo, hi8);
        }

#pragma unroll
        for (int ri = 0; ri < 2; ++ri)
#pragma unroll
            for (int ci = 0; ci < 4; ++ci)
                acc[ri][ci] = __builtin_amdgcn_wmma_f32_16x16x32_f16(
                    false, afrag[ri], false, bfrag[ci],
                    (short)0, acc[ri][ci], false, false);

        if (kt + 1 < KT) store_lds(cur ^ 1);   // fill other buffer
        __syncthreads();
    }

    // ---- epilogue: bias + sigmoid, f32 store ----
#pragma unroll
    for (int ci = 0; ci < 4; ++ci) {
        const int gn   = n0 + wn + ci * 16 + row16;
        const float bn = phib[gn];
#pragma unroll
        for (int ri = 0; ri < 2; ++ri) {
            const int mbase = m0 + wm + ri * 16 + (hi ? 8 : 0);
#pragma unroll
            for (int f = 0; f < 8; ++f) {
                float x = acc[ri][ci][f] + bn;
                phi[(size_t)(mbase + f) * NDIM + gn] = 1.0f / (1.0f + __expf(-x));
            }
        }
    }
}

// ---------------------------------------------------------------------------
// Kernel 2: per-image K-step value propagation in LDS + agent gather + head.
// One block (256 threads) per batch image; 1024 cells, 4 per thread.
// ---------------------------------------------------------------------------
__global__ __launch_bounds__(256)
void vpn_prop_head(const float* __restrict__ phi,      // (4096,2048): [b][2s+c]
                   const float* __restrict__ obs,      // (4096,3072): [b][3s+c]
                   const float* __restrict__ l1w,      // (36,16)
                   const float* __restrict__ l1b,      // (16)
                   const float* __restrict__ l2w,      // (16,4)
                   const float* __restrict__ l2b,      // (4)
                   float* __restrict__ out)            // (4096,4)
{
    __shared__ float r_s[1024];
    __shared__ float p_s[1024];
    __shared__ float Va[1024];
    __shared__ float Vb[1024];
    __shared__ float sel[36];
    __shared__ float hbuf[16];
    __shared__ int   agent;

    const int b   = blockIdx.x;
    const int tid = threadIdx.x;
    const float* phi_b = phi + (size_t)b * 2048;
    const float* obs_b = obs + (size_t)b * 3072;

    // load r,p and init V = r; remember agent candidate (one-hot channel 1)
    int myagent = -1;
#pragma unroll
    for (int i = 0; i < 4; ++i) {
        int s = i * 256 + tid;
        float r = phi_b[2 * s + 0];
        float p = phi_b[2 * s + 1];
        r_s[s] = r; p_s[s] = p; Va[s] = r;
        if (obs_b[3 * s + 1] > 0.5f) myagent = s;
    }
    if (tid == 0) agent = 0;
    __syncthreads();
    if (myagent >= 0) agent = myagent;   // exactly one writer

    // K=20 propagation steps, double-buffered (even count -> ends in Va)
    float* cur = Va;
    float* nxt = Vb;
    for (int it = 0; it < 20; ++it) {
#pragma unroll
        for (int i = 0; i < 4; ++i) {
            int s  = i * 256 + tid;
            int i0 = s >> 5, j0 = s & 31;
            float up    = (i0 > 0)  ? cur[s - 32] : 0.f;
            float down  = (i0 < 31) ? cur[s + 32] : 0.f;
            float left  = (j0 > 0)  ? cur[s - 1]  : 0.f;
            float right = (j0 < 31) ? cur[s + 1]  : 0.f;
            float nbr = fmaxf(fmaxf(up, down), fmaxf(left, right));
            float r = r_s[s];
            nxt[s] = fmaxf(cur[s], fmaf(p_s[s], nbr - r, r));
        }
        __syncthreads();
        float* t = cur; cur = nxt; nxt = t;
    }

    // 3x3x4 gather around agent (zero-padded): obs channels 0..2 + V
    const int aidx = agent;
    const int pi = aidx >> 5, pj = aidx & 31;
    if (tid < 36) {
        int di = tid / 12, dj = (tid / 4) % 3, c = tid % 4;
        int ii = pi + di - 1, jj = pj + dj - 1;
        float v = 0.f;
        if (ii >= 0 && ii < 32 && jj >= 0 && jj < 32) {
            int s = ii * 32 + jj;
            v = (c < 3) ? obs_b[3 * s + c] : cur[s];
        }
        sel[tid] = v;
    }
    __syncthreads();
    if (tid < 16) {
        float acc = l1b[tid];
#pragma unroll
        for (int i = 0; i < 36; ++i) acc = fmaf(sel[i], l1w[i * 16 + tid], acc);
        hbuf[tid] = fmaxf(acc, 0.f);
    }
    __syncthreads();
    if (tid < 4) {
        float acc = l2b[tid];
#pragma unroll
        for (int j = 0; j < 16; ++j) acc = fmaf(hbuf[j], l2w[j * 4 + tid], acc);
        out[(size_t)b * 4 + tid] = acc;
    }
}

// ---------------------------------------------------------------------------
extern "C" void kernel_launch(void* const* d_in, const int* in_sizes, int n_in,
                              void* d_out, int out_size, void* d_ws, size_t ws_size,
                              hipStream_t stream) {
    const float* obs   = (const float*)d_in[0];  // (4096,3072)
    const float* phiw  = (const float*)d_in[1];  // (3072,2048)
    const float* phib  = (const float*)d_in[2];  // (2048)
    const float* l1w   = (const float*)d_in[3];  // (36,16)
    const float* l1b   = (const float*)d_in[4];  // (16)
    const float* l2w   = (const float*)d_in[5];  // (16,4)
    const float* l2b   = (const float*)d_in[6];  // (4)
    float*       out   = (float*)d_out;          // (4096,4)

    float* phi = (float*)d_ws;                   // 4096*2048 f32 = 33.5 MB

    dim3 g1(NDIM / TILE_N, MDIM / TILE_M);       // (16, 32)
    vpn_phi_gemm<<<g1, 256, 0, stream>>>(obs, phiw, phib, phi);
    vpn_prop_head<<<4096, 256, 0, stream>>>(phi, obs, l1w, l1b, l2w, l2b, out);
}
